// NTXentLoss_55336358642715
// MI455X (gfx1250) — matrix operand order repaired and checked by reading, not compiled, not run on hardware
//
#include <hip/hip_runtime.h>
#include <hip/hip_bf16.h>

// NT-Xent loss for MI455X (gfx1250, wave32, WMMA).
//   B=4096, D=128, N=2B=8192, T=0.5  ->  sim scale = 2.0
// Compute-bound (17.2 GFLOP GEMM vs 4 MB input) -> v_wmma_f32_16x16x32_f16.
// Since vectors are unit-norm, sim <= 1/T = 2, so logsumexp uses the fixed
// shift 2: lse = 2 + log(sum_{j!=i} exp(sim_ij - 2)). No online max needed.
//
// Round-3: software pipelining across column tiles. B fragments are
// double-buffered (ping-pong); tile nt+1's 8x global_load_b128 are issued
// before tile nt's 8 WMMAs + exp epilogue, so every s_wait_loadcnt is
// covered by a full tile of matrix work instead of raw L0/L2 latency.

typedef __attribute__((ext_vector_type(16))) _Float16 v16h;
typedef __attribute__((ext_vector_type(8)))  _Float16 v8h;
typedef __attribute__((ext_vector_type(4)))  _Float16 h4;
typedef __attribute__((ext_vector_type(8)))  float    v8f;
typedef __attribute__((ext_vector_type(4)))  float    f4;

#define B_ROWS 4096
#define N_ROWS 8192
#define DIM    128
#define NCHUNKS 32                    // column chunks over N
#define CHUNK   (N_ROWS / NCHUNKS)    // 256 columns per chunk
#define NTILES  (CHUNK / 16)          // 16 column tiles per chunk

// ---------------------------------------------------------------- init ----
__global__ void ntxent_init_kernel(float* __restrict__ rowSum,
                                   float* __restrict__ out) {
    int i = blockIdx.x * blockDim.x + threadIdx.x;
    if (i < N_ROWS) rowSum[i] = 0.0f;
    if (i == 0)     out[0]    = 0.0f;
}

// ----------------------------------------------------------- normalize ----
// One wave32 per row: 32 lanes x float4 = 128 elements. Writes normalized
// f32 (for exact positive-pair dot) and f16 (WMMA operand matrix).
__global__ __launch_bounds__(256) void ntxent_normalize_kernel(
    const float* __restrict__ zi, const float* __restrict__ zj,
    _Float16* __restrict__ zh, float* __restrict__ zf) {
    const int lane = threadIdx.x & 31;
    const int row  = blockIdx.x * 8 + (threadIdx.x >> 5);   // 8 waves/block

    const float* src = (row < B_ROWS) ? (zi + (size_t)row * DIM)
                                      : (zj + (size_t)(row - B_ROWS) * DIM);
    f4 v = ((const f4*)src)[lane];
    float ss = v.x * v.x + v.y * v.y + v.z * v.z + v.w * v.w;
    ss += __shfl_xor(ss, 16, 32);
    ss += __shfl_xor(ss,  8, 32);
    ss += __shfl_xor(ss,  4, 32);
    ss += __shfl_xor(ss,  2, 32);
    ss += __shfl_xor(ss,  1, 32);
    const float s = 1.0f / fmaxf(sqrtf(ss), 1e-12f);   // matches F.normalize eps

    f4 nv = v * s;
    ((f4*)(zf + (size_t)row * DIM))[lane] = nv;

    h4 hv;
    hv.x = (_Float16)nv.x; hv.y = (_Float16)nv.y;
    hv.z = (_Float16)nv.z; hv.w = (_Float16)nv.w;
    ((h4*)(zh + (size_t)row * DIM))[lane] = hv;
}

// ------------------------------------------------- sim tile + exp-sum -----
// Block = 128 threads = 4 waves. grid = (N/128 row blocks, 32 chunks).
// Each wave: 32 rows (two 16-row A tile sets, loaded once) x one shared
// 256-column chunk (same chunk for all 4 waves -> WGP$-resident B stream).
// Software-pipelined over 16 column tiles with ping-pong B buffers.
__global__ __launch_bounds__(128) void ntxent_simlse_kernel(
    const _Float16* __restrict__ zh, float* __restrict__ rowSum) {
    const int lane  = threadIdx.x & 31;
    const int wave  = threadIdx.x >> 5;
    const int mBase = (blockIdx.x * 4 + wave) * 32;   // 32 rows per wave
    const int nStart = blockIdx.y * CHUNK;            // chunk shared by block

    const int lo16 = lane & 15;
    const int hi   = lane >> 4;

    // A-matrix 16x32 f16 layout (ISA 7.12.2): lane = M row; lanes 0-15 hold
    // K = ks*32 + {0..7, 16..23}, lanes 16-31 hold K = ks*32 + {8..15, 24..31}.
    v16h afrag[2][4];
#pragma unroll
    for (int mt = 0; mt < 2; ++mt) {
        const int arow = mBase + mt * 16 + lo16;
#pragma unroll
        for (int ks = 0; ks < 4; ++ks) {
            const _Float16* ap = zh + (size_t)arow * DIM + ks * 32 + hi * 8;
            v8h alo = *(const v8h*)(ap);        // first 8 K values for this lane
            v8h ahi = *(const v8h*)(ap + 16);   // second 8 K values (+16)
            afrag[mt][ks] = __builtin_shufflevector(alo, ahi,
                0, 1, 2, 3, 4, 5, 6, 7, 8, 9, 10, 11, 12, 13, 14, 15);
        }
    }

    float psum0[8], psum1[8];
#pragma unroll
    for (int r = 0; r < 8; ++r) { psum0[r] = 0.0f; psum1[r] = 0.0f; }

    const int rb0 = mBase + hi * 8;        // C/D: VGPR r -> row rb + r
    const int rb1 = mBase + 16 + hi * 8;

    // B-matrix 32x16: same K order as A: lanes 0-15 hold K {0..7, 16..23},
    // lanes 16-31 hold K {8..15, 24..31} -> two 16B loads per K-step.
    auto loadB = [&](int nt, v16h* dst) {
        const _Float16* bp =
            zh + (size_t)(nStart + nt * 16 + lo16) * DIM + hi * 8;
#pragma unroll
        for (int ks = 0; ks < 4; ++ks) {
            v8h blo = *(const v8h*)(bp + ks * 32);
            v8h bhi = *(const v8h*)(bp + ks * 32 + 16);
            dst[ks] = __builtin_shufflevector(blo, bhi,
                0, 1, 2, 3, 4, 5, 6, 7, 8, 9, 10, 11, 12, 13, 14, 15);
        }
    };

    auto compute = [&](int nt, const v16h* bf) {
        v8f c0 = {}, c1 = {};
#pragma unroll
        for (int ks = 0; ks < 4; ++ks) {
            c0 = __builtin_amdgcn_wmma_f32_16x16x32_f16(
                     false, afrag[0][ks], false, bf[ks], (short)0, c0, false, false);
            c1 = __builtin_amdgcn_wmma_f32_16x16x32_f16(
                     false, afrag[1][ks], false, bf[ks], (short)0, c1, false, false);
        }
        const int col = nStart + nt * 16 + lo16;
#pragma unroll
        for (int r = 0; r < 8; ++r) {
            float e0 = __expf(c0[r] * 2.0f - 2.0f);  // exp(sim/T - 2), <= 1
            float e1 = __expf(c1[r] * 2.0f - 2.0f);
            if (rb0 + r == col) e0 = 0.0f;           // mask diagonal (self)
            if (rb1 + r == col) e1 = 0.0f;
            psum0[r] += e0;
            psum1[r] += e1;
        }
    };

    v16h bfA[4], bfB[4];
    loadB(0, bfA);
#pragma unroll 2
    for (int nt = 0; nt < NTILES; nt += 2) {
        loadB(nt + 1, bfB);                 // prefetch tile nt+1
        compute(nt, bfA);                   // 8 WMMAs + 16 exp cover bfB loads
        const int pf = (nt + 2 < NTILES) ? nt + 2 : NTILES - 1;  // stay in-bounds
        loadB(pf, bfA);                     // prefetch tile nt+2
        compute(nt + 1, bfB);               // covers bfA loads
    }

    // Reduce each row's 16 column-lane partials within the half-wave.
#pragma unroll
    for (int r = 0; r < 8; ++r) {
        float s0 = psum0[r], s1 = psum1[r];
        s0 += __shfl_xor(s0, 1, 16);  s1 += __shfl_xor(s1, 1, 16);
        s0 += __shfl_xor(s0, 2, 16);  s1 += __shfl_xor(s1, 2, 16);
        s0 += __shfl_xor(s0, 4, 16);  s1 += __shfl_xor(s1, 4, 16);
        s0 += __shfl_xor(s0, 8, 16);  s1 += __shfl_xor(s1, 8, 16);
        psum0[r] = s0; psum1[r] = s1;
    }
    if (lo16 == 0) {
#pragma unroll
        for (int r = 0; r < 8; ++r) {
            atomicAdd(&rowSum[rb0 + r], psum0[r]);   // global_atomic_add_f32
            atomicAdd(&rowSum[rb1 + r], psum1[r]);
        }
    }
}

// ------------------------------------------------------------ finalize ----
// One thread per row: exact f32 positive-pair dot, lse = 2 + log(rowSum),
// wave-reduced mean accumulated into out.
__global__ __launch_bounds__(128) void ntxent_finalize_kernel(
    const float* __restrict__ zf, const float* __restrict__ rowSum,
    float* __restrict__ out) {
    const int i = blockIdx.x * 128 + threadIdx.x;      // 0..N-1
    const int j = (i + B_ROWS) & (N_ROWS - 1);         // (i + B) mod N

    const f4* a = (const f4*)(zf + (size_t)i * DIM);
    const f4* b = (const f4*)(zf + (size_t)j * DIM);
    float d = 0.0f;
#pragma unroll 8
    for (int k = 0; k < DIM / 4; ++k) {
        f4 x = a[k], y = b[k];
        d += x.x * y.x + x.y * y.y + x.z * y.z + x.w * y.w;
    }
    float li = -(d * 2.0f) + 2.0f + __logf(rowSum[i]); // -pos + lse

    li += __shfl_xor(li, 16, 32);
    li += __shfl_xor(li,  8, 32);
    li += __shfl_xor(li,  4, 32);
    li += __shfl_xor(li,  2, 32);
    li += __shfl_xor(li,  1, 32);
    if ((threadIdx.x & 31) == 0)
        atomicAdd(out, li * (1.0f / (float)N_ROWS));
}

// -------------------------------------------------------------- launch ----
extern "C" void kernel_launch(void* const* d_in, const int* in_sizes, int n_in,
                              void* d_out, int out_size, void* d_ws, size_t ws_size,
                              hipStream_t stream) {
    const float* zi = (const float*)d_in[0];   // [B, D] f32
    const float* zj = (const float*)d_in[1];   // [B, D] f32
    float* out = (float*)d_out;                // scalar f32

    // Workspace layout: zh f16 [N*D] (2 MB) | zf f32 [N*D] (4 MB) | rowSum f32 [N]
    char* ws = (char*)d_ws;
    _Float16* zh = (_Float16*)ws;
    float* zf     = (float*)(ws + (size_t)N_ROWS * DIM * sizeof(_Float16));
    float* rowSum = (float*)(ws + (size_t)N_ROWS * DIM * (sizeof(_Float16) + sizeof(float)));

    ntxent_init_kernel<<<(N_ROWS + 255) / 256, 256, 0, stream>>>(rowSum, out);
    ntxent_normalize_kernel<<<N_ROWS / 8, 256, 0, stream>>>(zi, zj, zh, zf);
    dim3 g2(N_ROWS / 128, NCHUNKS);    // 64 row blocks (4 waves x 32 rows) x 32 chunks
    ntxent_simlse_kernel<<<g2, 128, 0, stream>>>(zh, rowSum);
    ntxent_finalize_kernel<<<N_ROWS / 128, 128, 0, stream>>>(zf, rowSum, out);
}
